// BasicPBC_55155970015851
// MI455X (gfx1250) — matrix operand order repaired and checked by reading, not compiled, not run on hardware
//
#include <hip/hip_runtime.h>

// ---------------------------------------------------------------------------
// SuperGlue-style attentional GNN for MI455X (gfx1250).
// All GEMM-shaped math -> v_wmma_f32_16x16x32_f16 (f16 operands, f32 acc).
// GEMM: 2x2 register tiling per wave (4 WMMAs per k-step, halved LDS traffic
// per WMMA); all fragments are contiguous 16B LDS reads fed by coalesced
// b128 global loads. f32->f16 conversion done once per tensor.
// B=2, C=256, N=2048, H=4, dh=64, layers: self, cross, self, cross.
// ---------------------------------------------------------------------------

typedef __attribute__((ext_vector_type(16))) _Float16 v16h;
typedef __attribute__((ext_vector_type(8)))  _Float16 v8h;
typedef __attribute__((ext_vector_type(8)))  float    v8f;

#define FDIM   256
#define NPTS   2048
#define NHEADS 4
#define DHEAD  64
#define BATCH  2

__device__ __forceinline__ v16h cat8(v8h lo, v8h hi) {
  v16h r;
#pragma unroll
  for (int i = 0; i < 8; ++i) { r[i] = lo[i]; r[8 + i] = hi[i]; }
  return r;
}

// ---------------------------------------------------------------------------
// f32 -> f16 elementwise (n must be a multiple of 256)
// ---------------------------------------------------------------------------
__global__ void __launch_bounds__(256) cvt_f16_kernel(
    const float* __restrict__ in, _Float16* __restrict__ out)
{
  const long long i = (long long)blockIdx.x * 256 + threadIdx.x;
  out[i] = (_Float16)in[i];
}

// ---------------------------------------------------------------------------
// GEMM: Y[b][m][n] = sum_k W[m][k] * X[b][k][n] + bias[m] (+ resid)
// W, X f16; optional f32 output Yf and/or f16 output Yh.
// Block = 256 thr (8 waves), block tile 64(M) x 128(N), K step 32.
// Each wave owns a 32x32 register tile = 2x2 WMMA tiles (4 WMMA / k-step).
// ---------------------------------------------------------------------------
__global__ void __launch_bounds__(256) wmma_gemm_kernel(
    const _Float16* __restrict__ Wh, const _Float16* __restrict__ Xh,
    const float* __restrict__ bias, const float* __restrict__ resid,
    float* __restrict__ Yf, _Float16* __restrict__ Yh,
    int M, int K, int Ncols, long long yfStride, long long yhStride)
{
  __shared__ _Float16 Ws[64][32];       // [m][k]
  __shared__ _Float16 XsT[128][40];     // [n][k] (padded rows, 16B aligned)

  const int t    = threadIdx.x;
  const int wave = t >> 5;
  const int lane = t & 31;
  const int g    = lane >> 4;
  const int c16  = lane & 15;
  const int wm   = wave >> 2;           // 0..1  (32-row slab)
  const int wn   = wave & 3;            // 0..3  (32-col slab)

  const int b   = blockIdx.z;
  const int m00 = blockIdx.y << 6;      // 64 rows / block
  const int n00 = blockIdx.x << 7;      // 128 cols / block

  const _Float16* Xb = Xh + (long long)b * K * Ncols;

  v8f acc[2][2];
  const v8f vzero = {0.f, 0.f, 0.f, 0.f, 0.f, 0.f, 0.f, 0.f};
#pragma unroll
  for (int mi = 0; mi < 2; ++mi)
#pragma unroll
    for (int ni = 0; ni < 2; ++ni) acc[mi][ni] = vzero;

  // cooperative-load coordinates
  const int wlr = t >> 2;               // 0..63  W row
  const int wlc = (t & 3) << 3;         // 0..24  W col group (8 halves)
  const int xlk = t >> 4;               // 0..15  X k row (and +16)
  const int xln = (t & 15) << 3;        // 0..120 X n col group (8 halves)

  for (int k0 = 0; k0 < K; k0 += 32) {
    __syncthreads();
    // W tile 64x32: one 16B load per thread
    *(v8h*)&Ws[wlr][wlc] = *(const v8h*)&Wh[(long long)(m00 + wlr) * K + k0 + wlc];
    // X tile 32x128: two 16B loads per thread, transposed into LDS
    const v8h x0 = *(const v8h*)&Xb[(long long)(k0 + xlk) * Ncols + n00 + xln];
    const v8h x1 = *(const v8h*)&Xb[(long long)(k0 + 16 + xlk) * Ncols + n00 + xln];
#pragma unroll
    for (int j = 0; j < 8; ++j) {
      XsT[xln + j][xlk]      = x0[j];
      XsT[xln + j][16 + xlk] = x1[j];
    }
    __syncthreads();

    // A fragments: K slots {8g..8g+7, 16+8g..} -> two 16B reads each
    v16h af[2];
#pragma unroll
    for (int mi = 0; mi < 2; ++mi) {
      const int ml = (wm << 5) + (mi << 4) + c16;
      af[mi] = cat8(*(const v8h*)&Ws[ml][g << 3],
                    *(const v8h*)&Ws[ml][16 + (g << 3)]);
    }
    // B fragments: K slots {8g..8g+7, 16+8g..} -> two 16B reads each
    v16h bf[2];
#pragma unroll
    for (int ni = 0; ni < 2; ++ni) {
      const int nl = (wn << 5) + (ni << 4) + c16;
      bf[ni] = cat8(*(const v8h*)&XsT[nl][g << 3],
                    *(const v8h*)&XsT[nl][16 + (g << 3)]);
    }
    // 4 WMMAs per k-step
#pragma unroll
    for (int mi = 0; mi < 2; ++mi)
#pragma unroll
      for (int ni = 0; ni < 2; ++ni)
        acc[mi][ni] = __builtin_amdgcn_wmma_f32_16x16x32_f16(
            false, af[mi], false, bf[ni], (short)0, acc[mi][ni], false, false);
  }

  const float* Rb = resid ? (resid + (long long)b * M * Ncols) : nullptr;
#pragma unroll
  for (int mi = 0; mi < 2; ++mi) {
#pragma unroll
    for (int ni = 0; ni < 2; ++ni) {
#pragma unroll
      for (int j = 0; j < 8; ++j) {
        const int m = m00 + (wm << 5) + (mi << 4) + j + (g << 3);
        const int n = n00 + (wn << 5) + (ni << 4) + c16;
        float vv = acc[mi][ni][j] + (bias ? bias[m] : 0.f);
        if (Rb) vv += Rb[(long long)m * Ncols + n];
        if (Yf) Yf[(long long)b * yfStride + (long long)m * Ncols + n] = vv;
        if (Yh) Yh[(long long)b * yhStride + (long long)m * Ncols + n] = (_Float16)vv;
      }
    }
  }
}

// ---------------------------------------------------------------------------
// Flash attention: q,k,v are f16 [B, DHEAD, NHEADS, NPTS] (channel c = d*H+h).
// 4 waves/block share (b,h); K/V tiles staged in LDS with wide loads each
// 32-column chunk. Online softmax; 8 WMMAs per chunk. Output msg f16 [B,C,N].
// ---------------------------------------------------------------------------
__global__ void __launch_bounds__(128) flash_attn_kernel(
    const _Float16* __restrict__ q, const _Float16* __restrict__ k,
    const _Float16* __restrict__ v, _Float16* __restrict__ msg)
{
  __shared__ _Float16 pst[4][16][32];   // per-wave P tile (n x m)
  __shared__ _Float16 KsT[32][72];      // K tile transposed: [m][d]
  __shared__ _Float16 Vs[64][40];       // V tile: [d][m]

  const int t    = threadIdx.x;
  const int wave = t >> 5;
  const int lane = t & 31;
  const int g    = lane >> 4;
  const int c16  = lane & 15;
  const int h    = blockIdx.y;
  const int b    = blockIdx.z;
  const int n0   = ((blockIdx.x << 2) + wave) << 4;

  float mrow[8], lrow[8];
  v8f   oacc[4];
  const v8f vzero = {0.f, 0.f, 0.f, 0.f, 0.f, 0.f, 0.f, 0.f};
#pragma unroll
  for (int j = 0; j < 8; ++j) { mrow[j] = -1e30f; lrow[j] = 0.f; }
#pragma unroll
  for (int dc = 0; dc < 4; ++dc) oacc[dc] = vzero;

  // Q^T fragments (rows = n, K = d in [0,32) and [32,64)), loaded once.
  v16h aq0, aq1;
#pragma unroll
  for (int i = 0; i < 16; ++i) {
    const int r  = i >> 1;
    const int dd = (g << 4) + i; // kmapA
    aq0[i] = q[(((long long)b * DHEAD + dd)      * NHEADS + h) * NPTS + n0 + c16];
    aq1[i] = q[(((long long)b * DHEAD + 32 + dd) * NHEADS + h) * NPTS + n0 + c16];
  }

  // cooperative-load coordinates (128 threads): d = t/2, m-half = (t&1)*16
  const int ld = t >> 1;
  const int lm = (t & 1) << 4;

  for (int m0 = 0; m0 < NPTS; m0 += 32) {
    __syncthreads();
    // K tile -> transposed LDS; V tile -> LDS (all 16B global loads)
    {
      const long long kb = (((long long)b * DHEAD + ld) * NHEADS + h) * NPTS + m0 + lm;
      const v8h k0v = *(const v8h*)&k[kb];
      const v8h k1v = *(const v8h*)&k[kb + 8];
#pragma unroll
      for (int j = 0; j < 8; ++j) { KsT[lm + j][ld] = k0v[j]; KsT[lm + 8 + j][ld] = k1v[j]; }
      *(v8h*)&Vs[ld][lm]     = *(const v8h*)&v[kb];
      *(v8h*)&Vs[ld][lm + 8] = *(const v8h*)&v[kb + 8];
    }
    __syncthreads();

    // ---- scores: two 16x16 tiles; B fragments from KsT (contiguous d)
    v8f s1 = vzero, s2 = vzero;
    {
      const v16h bk1 = cat8(*(const v8h*)&KsT[c16][g << 4],
                            *(const v8h*)&KsT[c16][(g << 4) + 8]);
      const v16h bk2 = cat8(*(const v8h*)&KsT[16 + c16][g << 4],
                            *(const v8h*)&KsT[16 + c16][(g << 4) + 8]);
      s1 = __builtin_amdgcn_wmma_f32_16x16x32_f16(false, aq0, false, bk1, (short)0, s1, false, false);
      s2 = __builtin_amdgcn_wmma_f32_16x16x32_f16(false, aq0, false, bk2, (short)0, s2, false, false);
    }
    {
      const v16h bk1 = cat8(*(const v8h*)&KsT[c16][32 + (g << 4)],
                            *(const v8h*)&KsT[c16][32 + (g << 4) + 8]);
      const v16h bk2 = cat8(*(const v8h*)&KsT[16 + c16][32 + (g << 4)],
                            *(const v8h*)&KsT[16 + c16][32 + (g << 4) + 8]);
      s1 = __builtin_amdgcn_wmma_f32_16x16x32_f16(false, aq1, false, bk1, (short)0, s1, false, false);
      s2 = __builtin_amdgcn_wmma_f32_16x16x32_f16(false, aq1, false, bk2, (short)0, s2, false, false);
    }

    // ---- online softmax (row = j + 8g, col = lane within half)
#pragma unroll
    for (int j = 0; j < 8; ++j) {
      const float x1 = s1[j] * 0.125f;  // 1/sqrt(64)
      const float x2 = s2[j] * 0.125f;
      float tm = fmaxf(x1, x2);
#pragma unroll
      for (int mk = 8; mk >= 1; mk >>= 1) tm = fmaxf(tm, __shfl_xor(tm, mk, 32));
      const float mnew  = fmaxf(mrow[j], tm);
      const float scale = __expf(mrow[j] - mnew);
      const float e1 = __expf(x1 - mnew);
      const float e2 = __expf(x2 - mnew);
      float rs = e1 + e2;
#pragma unroll
      for (int mk = 8; mk >= 1; mk >>= 1) rs += __shfl_xor(rs, mk, 32);
      lrow[j] = lrow[j] * scale + rs;
      mrow[j] = mnew;
#pragma unroll
      for (int dc = 0; dc < 4; ++dc) oacc[dc][j] *= scale;
      pst[wave][j + (g << 3)][c16]      = (_Float16)e1;
      pst[wave][j + (g << 3)][16 + c16] = (_Float16)e2;
    }
    __syncthreads();

    // ---- P (16n x 32m) x V^T (32m x 64d): 4 WMMAs over d chunks
    const v16h ap = cat8(*(const v8h*)&pst[wave][c16][g << 4],
                         *(const v8h*)&pst[wave][c16][(g << 4) + 8]);
#pragma unroll
    for (int dc = 0; dc < 4; ++dc) {
      const int dr = (dc << 4) + c16;
      const v16h bv = cat8(*(const v8h*)&Vs[dr][g << 4],
                           *(const v8h*)&Vs[dr][(g << 4) + 8]);
      oacc[dc] = __builtin_amdgcn_wmma_f32_16x16x32_f16(false, ap, false, bv,
                                                        (short)0, oacc[dc], false, false);
    }
  }

  // ---- normalize, scatter to [B, FDIM, NPTS] as f16 (c = d*H + h)
#pragma unroll
  for (int dc = 0; dc < 4; ++dc) {
    const int d = (dc << 4) + c16;
    const int c = d * NHEADS + h;
#pragma unroll
    for (int j = 0; j < 8; ++j) {
      const int n = n0 + j + (g << 3);
      msg[((long long)b * FDIM + c) * NPTS + n] = (_Float16)(oacc[dc][j] / lrow[j]);
    }
  }
}

// ---------------------------------------------------------------------------
// InstanceNorm1d (affine=False) + ReLU; f32 in, f16 out (same layout).
// One block per (b, channel) row.
// ---------------------------------------------------------------------------
__global__ void __launch_bounds__(256) inorm_relu_kernel(
    const float* __restrict__ hin, _Float16* __restrict__ hout, int ncols)
{
  __shared__ float ssum[256];
  __shared__ float ssq[256];
  const float* p = hin + (long long)blockIdx.x * ncols;
  _Float16*    o = hout + (long long)blockIdx.x * ncols;
  float s = 0.f, s2 = 0.f;
  for (int i = threadIdx.x; i < ncols; i += 256) {
    const float v = p[i];
    s += v; s2 += v * v;
  }
  ssum[threadIdx.x] = s; ssq[threadIdx.x] = s2;
  __syncthreads();
  for (int off = 128; off > 0; off >>= 1) {
    if (threadIdx.x < off) {
      ssum[threadIdx.x] += ssum[threadIdx.x + off];
      ssq[threadIdx.x]  += ssq[threadIdx.x + off];
    }
    __syncthreads();
  }
  const float mean = ssum[0] / (float)ncols;
  const float var  = fmaxf(ssq[0] / (float)ncols - mean * mean, 0.f);
  const float inv  = rsqrtf(var + 1e-5f);
  for (int i = threadIdx.x; i < ncols; i += 256) {
    const float v = (p[i] - mean) * inv;
    o[i] = (_Float16)(v > 0.f ? v : 0.f);
  }
}

// ---------------------------------------------------------------------------
// Host orchestration
// ---------------------------------------------------------------------------
extern "C" void kernel_launch(void* const* d_in, const int* in_sizes, int n_in,
                              void* d_out, int out_size, void* d_ws, size_t ws_size,
                              hipStream_t stream) {
  (void)in_sizes; (void)n_in; (void)out_size; (void)ws_size;

  const float* desc0 = (const float*)d_in[0];
  const float* desc1 = (const float*)d_in[1];
  auto P = [&](int layer, int idx) -> const float* {
    return (const float*)d_in[2 + layer * 12 + idx];
  };

  const long long SZ = (long long)BATCH * FDIM * NPTS;   // 1,048,576 elements

  // f32 region
  float* f   = (float*)d_ws;
  float* d0  = f;             // current desc0
  float* d1  = d0 + SZ;
  float* nd0 = d1 + SZ;
  float* nd1 = nd0 + SZ;
  float* h1  = nd1 + SZ;      // [B,512,N] pre-norm hidden (2*SZ)
  // f16 region
  _Float16* hr   = (_Float16*)(h1 + 2 * SZ);
  _Float16* d0h  = hr;                // f16 copies of current descs
  _Float16* d1h  = d0h + SZ;
  _Float16* qh   = d1h + SZ;
  _Float16* kh   = qh + SZ;
  _Float16* vh   = kh + SZ;
  _Float16* atth = vh + SZ;           // attention output, f16
  _Float16* hcat = atth + SZ;         // [B,512,N] concat, f16 (2*SZ)
  _Float16* h1h  = hcat + 2 * SZ;     // normalized hidden, f16 (2*SZ)
  _Float16* wqh  = h1h + 2 * SZ;      // per-layer f16 weights
  _Float16* wkh  = wqh + FDIM * FDIM;
  _Float16* wvh  = wkh + FDIM * FDIM;
  _Float16* wmh  = wvh + FDIM * FDIM;
  _Float16* w1h  = wmh + FDIM * FDIM;
  _Float16* w2h  = w1h + 4 * FDIM * FDIM;

  hipMemcpyAsync(d0, desc0, SZ * sizeof(float), hipMemcpyDeviceToDevice, stream);
  hipMemcpyAsync(d1, desc1, SZ * sizeof(float), hipMemcpyDeviceToDevice, stream);

  auto cvt = [&](const float* src, _Float16* dst, long long n) {
    cvt_f16_kernel<<<(unsigned)(n / 256), 256, 0, stream>>>(src, dst);
  };
  auto gemm = [&](const _Float16* W, const _Float16* X, const float* bias,
                  const float* resid, float* Yf, _Float16* Yh,
                  int M, int K, long long yfStride, long long yhStride) {
    dim3 grid(NPTS / 128, M / 64, BATCH);
    wmma_gemm_kernel<<<grid, 256, 0, stream>>>(W, X, bias, resid, Yf, Yh,
                                               M, K, NPTS, yfStride, yhStride);
  };

  for (int L = 0; L < 4; ++L) {
    const bool cross = (L & 1);          // self, cross, self, cross

    // convert this layer's weights to f16
    cvt(P(L, 0), wqh, FDIM * FDIM);
    cvt(P(L, 2), wkh, FDIM * FDIM);
    cvt(P(L, 4), wvh, FDIM * FDIM);
    cvt(P(L, 6), wmh, FDIM * FDIM);
    cvt(P(L, 8), w1h, 4 * FDIM * FDIM);
    cvt(P(L, 10), w2h, 2 * FDIM * FDIM);
    // f16 copies of current descriptors
    cvt(d0, d0h, SZ);
    cvt(d1, d1h, SZ);

    for (int t = 0; t < 2; ++t) {
      const float*    x    = (t == 0) ? d0 : d1;
      const _Float16* xh   = (t == 0) ? d0h : d1h;
      const _Float16* srch = cross ? ((t == 0) ? d1h : d0h) : xh;
      float*          onew = (t == 0) ? nd0 : nd1;

      // QKV projections -> f16 outputs only
      gemm(wqh, xh,   P(L, 1), nullptr, nullptr, qh, FDIM, FDIM, 0, SZ / BATCH);
      gemm(wkh, srch, P(L, 3), nullptr, nullptr, kh, FDIM, FDIM, 0, SZ / BATCH);
      gemm(wvh, srch, P(L, 5), nullptr, nullptr, vh, FDIM, FDIM, 0, SZ / BATCH);

      // multi-head attention (flash, all WMMA)
      flash_attn_kernel<<<dim3((NPTS / 16) / 4, NHEADS, BATCH), 128, 0, stream>>>(
          qh, kh, vh, atth);

      // concat: top half = xh, bottom half = merge(att) (f16)
      for (int bb = 0; bb < BATCH; ++bb) {
        hipMemcpyAsync(hcat + (long long)bb * 2 * FDIM * NPTS,
                       xh + (long long)bb * FDIM * NPTS,
                       (size_t)FDIM * NPTS * sizeof(_Float16),
                       hipMemcpyDeviceToDevice, stream);
      }
      gemm(wmh, atth, P(L, 7), nullptr, nullptr, hcat + (long long)FDIM * NPTS,
           FDIM, FDIM, 0, (long long)2 * FDIM * NPTS);

      // MLP layer 1 (512x512) -> f32 h1 for normalization
      gemm(w1h, hcat, P(L, 9), nullptr, h1, nullptr, 2 * FDIM, 2 * FDIM,
           (long long)2 * FDIM * NPTS, 0);

      // instance norm + relu -> f16 h1h
      inorm_relu_kernel<<<BATCH * 2 * FDIM, 256, 0, stream>>>(h1, h1h, NPTS);

      // MLP layer 2 + residual: onew = x + W2*h1h + b2  (f32 out)
      gemm(w2h, h1h, P(L, 11), x, onew, nullptr, FDIM, 2 * FDIM,
           (long long)FDIM * NPTS, 0);
    }
    // ping-pong descriptors
    float* tmp;
    tmp = d0; d0 = nd0; nd0 = tmp;
    tmp = d1; d1 = nd1; nd1 = tmp;
  }

  hipMemcpyAsync(d_out, d0, SZ * sizeof(float), hipMemcpyDeviceToDevice, stream);
  hipMemcpyAsync((float*)d_out + SZ, d1, SZ * sizeof(float), hipMemcpyDeviceToDevice, stream);
}
